// FixedProductionSplatFlowAttention_43224550867632
// MI455X (gfx1250) — compile-verified, matrix operand order checked by reading, not run on hardware
//
#include <hip/hip_runtime.h>
#include <hip/hip_bf16.h>

typedef unsigned short u16;
typedef __attribute__((ext_vector_type(16))) __bf16 v16bf;
typedef __attribute__((ext_vector_type(8)))  float  v8f;

// TDM descriptor groups (amdgpu-toolchain 6-arg builtin form)
typedef __attribute__((ext_vector_type(4))) unsigned tdm_u4;
typedef __attribute__((ext_vector_type(8))) int      tdm_i8;
typedef __attribute__((ext_vector_type(4))) int      tdm_i4;

struct alignas(16) U4 { unsigned x, y, z, w; };
struct Frag32 { U4 lo, hi; };   // 32 bytes == v16bf

__device__ __forceinline__ u16 f32_to_bf16(float f) {
    unsigned u = __builtin_bit_cast(unsigned, f);
    unsigned r = u + 0x7FFFu + ((u >> 16) & 1u);
    return (u16)(r >> 16);
}
__device__ __forceinline__ float bf16_to_f32(u16 h) {
    unsigned u = ((unsigned)h) << 16;
    return __builtin_bit_cast(float, u);
}

// ---------------------------------------------------------------------------
// Issue one TDM 2D tile load: [64 rows x 32 cols] of 2-byte elements at
// (row0, kk) of a row-major matrix with leading dim ld, into LDS at lds_off.
// D# layout per CDNA5 ISA 8.3/8.4 (group0: count/lds/global/type,
// group1: data_size, tensor dims, tile dims, stride). Groups 2/3 disabled
// (2D tile), extra operands zero-filled.
// ---------------------------------------------------------------------------
__device__ __forceinline__ void tdm_issue_tile(const u16* gbase, int ld, int rows,
                                               unsigned lds_off, int kk, int row0) {
    unsigned long long ga = (unsigned long long)(gbase + (size_t)row0 * ld + kk);
    tdm_u4 g0;
    g0[0] = 1u;                                       // count=1, user mode
    g0[1] = lds_off;                                  // lds_addr (bytes)
    g0[2] = (unsigned)(ga & 0xFFFFFFFFu);             // global_addr[31:0]
    g0[3] = (unsigned)((ga >> 32) & 0x1FFFFFFu)       // global_addr[56:32]
          | (2u << 30);                               // type=2 ("image")
    tdm_i8 g1;
    unsigned uld = (unsigned)ld, urows = (unsigned)rows;
    g1[0] = (int)(1u << 16);                          // data_size=1 (2 bytes)
    g1[1] = (int)(uld << 16);                         // tensor_dim0[15:0]
    g1[2] = (int)((uld >> 16) | (urows << 16));       // tensor_dim0[31:16], dim1[15:0]
    g1[3] = (int)((urows >> 16) | (32u << 16));       // dim1[31:16], tile_dim0=32
    g1[4] = 64;                                       // tile_dim1=64, tile_dim2=0
    g1[5] = (int)uld;                                 // tensor_dim0_stride[31:0]
    g1[6] = 0;                                        // stride0[47:32]=0, stride1 lo=0
    g1[7] = 0;
    tdm_i4 z4 = {0, 0, 0, 0};
    tdm_i8 z8 = {0, 0, 0, 0, 0, 0, 0, 0};
    __builtin_amdgcn_tensor_load_to_lds(g0, g1, z4, z4, z8, 0);
}

// ---------------------------------------------------------------------------
// Tiled bf16 WMMA GEMM: C[M,N] (+batch) = A[M,Kd] @ BT[N,Kd]^T, all row-major.
// B is supplied pre-transposed, so A and B tiles have identical row-major
// [64][32] staging and identical per-lane fragment loads (2x ds_load_b128).
// Tiles are moved by the Tensor Data Mover (double-buffered): wave0 issues
// the next tile's TDM ops, s_wait_tensorcnt on the current tile, and the
// workgroup barrier publishes LDS to the other 3 waves.
// ---------------------------------------------------------------------------
template<int OUT_BF16>
__global__ __launch_bounds__(128)
void wmma_gemm(const u16* __restrict__ A, const u16* __restrict__ BT,
               void* __restrict__ Cv, int M, int N, int Kd,
               long long sA, long long sB, long long sC)
{
    __shared__ u16 As[2][64][32];
    __shared__ u16 Bs[2][64][32];

    const int t    = threadIdx.x;
    const int lane = t & 31;
    const int w    = t >> 5;
    const int wm   = (w >> 1) * 32;
    const int wn   = (w & 1) * 32;
    const int bm0  = blockIdx.y * 64;
    const int bn0  = blockIdx.x * 64;

    A  += (size_t)blockIdx.z * sA;
    BT += (size_t)blockIdx.z * sB;

    const unsigned asOff[2] = { (unsigned)(unsigned long long)(const void*)&As[0][0][0],
                                (unsigned)(unsigned long long)(const void*)&As[1][0][0] };
    const unsigned bsOff[2] = { (unsigned)(unsigned long long)(const void*)&Bs[0][0][0],
                                (unsigned)(unsigned long long)(const void*)&Bs[1][0][0] };

    v8f acc[2][2];
#pragma unroll
    for (int i = 0; i < 2; ++i)
#pragma unroll
        for (int j = 0; j < 2; ++j) acc[i][j] = (v8f){0,0,0,0,0,0,0,0};

    // fragment lane mapping (ISA 16-bit A/B layout, wave32)
    const int lrow  = lane & 15;
    const int khalf = (lane < 16) ? 0 : 8;

    const int T = Kd / 32;
    if (w == 0) {   // uniform per wave: EXEC within the wave stays full
        tdm_issue_tile(A,  Kd, M, asOff[0], 0, bm0);
        tdm_issue_tile(BT, Kd, N, bsOff[0], 0, bn0);
    }

    for (int s = 0; s < T; ++s) {
        const int buf = s & 1;
        if (w == 0) {
            if (s + 1 < T) {
                tdm_issue_tile(A,  Kd, M, asOff[buf ^ 1], (s + 1) * 32, bm0);
                tdm_issue_tile(BT, Kd, N, bsOff[buf ^ 1], (s + 1) * 32, bn0);
                __builtin_amdgcn_s_wait_tensorcnt(2);   // tile s complete
            } else {
                __builtin_amdgcn_s_wait_tensorcnt(0);
            }
        }
        __syncthreads();   // publish TDM-written LDS to all waves

        Frag32 fA[2], fB[2];
#pragma unroll
        for (int i = 0; i < 2; ++i) {
            fA[i].lo = *(const U4*)&As[buf][wm + i * 16 + lrow][khalf];
            fA[i].hi = *(const U4*)&As[buf][wm + i * 16 + lrow][khalf + 16];
            fB[i].lo = *(const U4*)&Bs[buf][wn + i * 16 + lrow][khalf];
            fB[i].hi = *(const U4*)&Bs[buf][wn + i * 16 + lrow][khalf + 16];
        }
#pragma unroll
        for (int i = 0; i < 2; ++i) {
            v16bf av = __builtin_bit_cast(v16bf, fA[i]);
#pragma unroll
            for (int j = 0; j < 2; ++j) {
                v16bf bv = __builtin_bit_cast(v16bf, fB[j]);
                acc[i][j] = __builtin_amdgcn_wmma_f32_16x16x32_bf16(
                    false, av, false, bv, (short)0, acc[i][j], false, false);
            }
        }
        __syncthreads();   // all waves done reading buf before it is re-issued
    }

    // C/D layout: VGPR r -> row = r + (lane<16 ? 0 : 8), col = lane&15
    const int rbase = (lane < 16) ? 0 : 8;
    const int ccol  = lane & 15;
    u16*   Cb = (u16*)Cv   + (size_t)blockIdx.z * sC;
    float* Cf = (float*)Cv + (size_t)blockIdx.z * sC;
#pragma unroll
    for (int i = 0; i < 2; ++i)
#pragma unroll
        for (int j = 0; j < 2; ++j) {
            int c = bn0 + wn + j * 16 + ccol;
#pragma unroll
            for (int r = 0; r < 8; ++r) {
                int row = bm0 + wm + i * 16 + rbase + r;
                float v = acc[i][j][r];
                if (OUT_BF16) Cb[(size_t)row * N + c] = f32_to_bf16(v);
                else          Cf[(size_t)row * N + c] = v;
            }
        }
}

// ---------------------------------------------------------------------------
// Elementwise / small kernels
// ---------------------------------------------------------------------------
__global__ void convert_bf16(const float* __restrict__ s, u16* __restrict__ d, int n) {
    int i = blockIdx.x * 256 + threadIdx.x;
    if (i < n) d[i] = f32_to_bf16(s[i]);
}

// batched transpose f32[b,r,c] -> bf16[b,c,r]
__global__ void transpose_f32_bf16(const float* __restrict__ s, u16* __restrict__ d,
                                   int batch, int rows, int cols) {
    int i = blockIdx.x * 256 + threadIdx.x;
    if (i < batch * rows * cols) {
        int b = i / (rows * cols), rem = i - b * rows * cols;
        int r = rem / cols, c = rem - r * cols;
        d[((size_t)b * cols + c) * rows + r] = f32_to_bf16(s[i]);
    }
}

// batched transpose bf16[b,r,c] -> bf16[b,c,r]
__global__ void transpose_u16(const u16* __restrict__ s, u16* __restrict__ d,
                              int batch, int rows, int cols) {
    int i = blockIdx.x * 256 + threadIdx.x;
    if (i < batch * rows * cols) {
        int b = i / (rows * cols), rem = i - b * rows * cols;
        int r = rem / cols, c = rem - r * cols;
        d[((size_t)b * cols + c) * rows + r] = s[i];
    }
}

// per-splat: p2[k] = |pos_k|^2 ; itv[k] = 0.5/(exp(2*ls)+eps)
__global__ void splat_prep(const float* __restrict__ pos, const float* __restrict__ ls,
                           float* __restrict__ p2, float* __restrict__ itv, int K, int D) {
    int k = threadIdx.x;
    if (k < K) {
        float s = 0.f;
        for (int d = 0; d < D; ++d) { float p = pos[k * D + d]; s += p * p; }
        p2[k] = s;
        float sc = __expf(ls[k]);
        itv[k] = 0.5f / (sc * sc + 1e-8f);
    }
}

// row norm of bf16 matrix [rows, D]
__global__ void rownorm(const u16* __restrict__ X, float* __restrict__ nrm, int rows, int D) {
    int r = blockIdx.x * 256 + threadIdx.x;
    if (r < rows) {
        const u16* p = X + (size_t)r * D;
        float s = 0.f;
        for (int d = 0; d < D; ++d) { float f = bf16_to_f32(p[d]); s += f * f; }
        nrm[r] = s;
    }
}

// g_q * amp -> gqa [rows, K] bf16
__global__ void gaussian_q(const float* __restrict__ xp, const float* __restrict__ nrm,
                           const float* __restrict__ p2, const float* __restrict__ itv,
                           const float* __restrict__ amp, u16* __restrict__ gqa, int n) {
    int i = blockIdx.x * 256 + threadIdx.x;
    if (i < n) {
        int r = i >> 6, k = i & 63;
        float d = fmaxf(nrm[r] - 2.f * xp[i] + p2[k], 0.f);
        float g = __expf(-d * itv[k]);
        gqa[i] = f32_to_bf16(g * amp[k]);
    }
}

// g_k written transposed: gkT[b, k, j] bf16
__global__ void gaussian_kT(const float* __restrict__ xp, const float* __restrict__ nrm,
                            const float* __restrict__ p2, const float* __restrict__ itv,
                            u16* __restrict__ gkT, int n, int S) {
    int i = blockIdx.x * 256 + threadIdx.x;
    if (i < n) {
        int r = i >> 6, k = i & 63;
        int b = r / S, j = r - b * S;
        float d = fmaxf(nrm[r] - 2.f * xp[i] + p2[k], 0.f);
        float g = __expf(-d * itv[k]);
        gkT[((size_t)b * 64 + k) * S + j] = f32_to_bf16(g);
    }
}

// column sums of g_k: s[b*K+k] = sum_j gkT[b,k,j]
__global__ void rowsum(const u16* __restrict__ gkT, float* __restrict__ s, int rows, int S) {
    int r = blockIdx.x * 256 + threadIdx.x;
    if (r < rows) {
        const u16* p = gkT + (size_t)r * S;
        float acc = 0.f;
        for (int j = 0; j < S; ++j) acc += bf16_to_f32(p[j]);
        s[r] = acc;
    }
}

// den[r] = sum_k gqa[r,k] * s[b,k]
__global__ void den_kernel(const u16* __restrict__ gqa, const float* __restrict__ s,
                           float* __restrict__ den, int rows, int S) {
    int r = blockIdx.x * 256 + threadIdx.x;
    if (r < rows) {
        int b = r / S;
        float acc = 0.f;
        for (int k = 0; k < 64; ++k)
            acc += bf16_to_f32(gqa[(size_t)r * 64 + k]) * s[b * 64 + k];
        den[r] = acc;
    }
}

// h = num / (den + eps), bf16
__global__ void normalize_h(const float* __restrict__ num, const float* __restrict__ den,
                            u16* __restrict__ h, int n, int D) {
    int i = blockIdx.x * 256 + threadIdx.x;
    if (i < n) h[i] = f32_to_bf16(num[i] / (den[i / D] + 1e-8f));
}

// ---------------------------------------------------------------------------
extern "C" void kernel_launch(void* const* d_in, const int* in_sizes, int n_in,
                              void* d_out, int out_size, void* d_ws, size_t ws_size,
                              hipStream_t stream) {
    const int B = 4, S = 2048, D = 768, K = 64;
    const int R = B * S;                 // 8192 rows
    const int nX = R * D;                // 6291456
    const int nW = D * D;                // 589824
    const int nXP = R * K;               // 524288

    const float* x   = (const float*)d_in[0];
    const float* Wq  = (const float*)d_in[1];
    const float* Wk  = (const float*)d_in[2];
    const float* Wv  = (const float*)d_in[3];
    const float* Wo  = (const float*)d_in[4];
    const float* pos = (const float*)d_in[5];
    const float* ls  = (const float*)d_in[6];
    const float* amp = (const float*)d_in[7];
    float* out = (float*)d_out;

    char* wsp = (char*)d_ws;
    auto alloc = [&](size_t bytes) -> char* {
        char* p = wsp; wsp += (bytes + 255) & ~(size_t)255; return p;
    };
    u16*   xb    = (u16*)alloc((size_t)nX * 2);
    u16*   wqT   = (u16*)alloc((size_t)nW * 2);   // Wq^T bf16
    u16*   wkT   = (u16*)alloc((size_t)nW * 2);
    u16*   wvT   = (u16*)alloc((size_t)nW * 2);
    u16*   woT   = (u16*)alloc((size_t)nW * 2);
    u16*   posB  = (u16*)alloc((size_t)K * D * 2);  // pos bf16 (already BT layout)
    float* p2    = (float*)alloc(K * 4);
    float* itv   = (float*)alloc(K * 4);
    u16*   Qb    = (u16*)alloc((size_t)nX * 2);
    u16*   Kb    = (u16*)alloc((size_t)nX * 2);
    u16*   Vb    = (u16*)alloc((size_t)nX * 2);
    u16*   VbT   = (u16*)alloc((size_t)nX * 2);     // [B, D, S]
    float* qn    = (float*)alloc(R * 4);
    float* kn    = (float*)alloc(R * 4);
    float* xpq   = (float*)alloc((size_t)nXP * 4);
    float* xpk   = (float*)alloc((size_t)nXP * 4);
    u16*   gqa   = (u16*)alloc((size_t)nXP * 2);
    u16*   gkT   = (u16*)alloc((size_t)nXP * 2);
    float* scol  = (float*)alloc(B * K * 4);
    float* Mf    = (float*)alloc((size_t)B * K * D * 4);
    u16*   MbT   = (u16*)alloc((size_t)B * K * D * 2);  // [B, D, K]
    float* Nmat  = (float*)alloc((size_t)nX * 4);
    float* den   = (float*)alloc(R * 4);
    u16*   hb    = (u16*)alloc((size_t)nX * 2);

    auto blk = [](int n) { return dim3((n + 255) / 256); };

    // conversions + pre-transposed B operands
    convert_bf16<<<blk(nX), 256, 0, stream>>>(x, xb, nX);
    transpose_f32_bf16<<<blk(nW), 256, 0, stream>>>(Wq, wqT, 1, D, D);
    transpose_f32_bf16<<<blk(nW), 256, 0, stream>>>(Wk, wkT, 1, D, D);
    transpose_f32_bf16<<<blk(nW), 256, 0, stream>>>(Wv, wvT, 1, D, D);
    transpose_f32_bf16<<<blk(nW), 256, 0, stream>>>(Wo, woT, 1, D, D);
    convert_bf16<<<blk(K * D), 256, 0, stream>>>(pos, posB, K * D);
    splat_prep<<<1, 64, 0, stream>>>(pos, ls, p2, itv, K, D);

    // Q, K, V projections: [8192,768] = [8192,768] @ [768,768], bf16 out
    dim3 gProj(D / 64, R / 64, 1);
    wmma_gemm<1><<<gProj, 128, 0, stream>>>(xb, wqT, Qb, R, D, D, 0, 0, 0);
    wmma_gemm<1><<<gProj, 128, 0, stream>>>(xb, wkT, Kb, R, D, D, 0, 0, 0);
    wmma_gemm<1><<<gProj, 128, 0, stream>>>(xb, wvT, Vb, R, D, D, 0, 0, 0);

    rownorm<<<blk(R), 256, 0, stream>>>(Qb, qn, R, D);
    rownorm<<<blk(R), 256, 0, stream>>>(Kb, kn, R, D);
    transpose_u16<<<blk(nX), 256, 0, stream>>>(Vb, VbT, B, S, D);

    // q@pos^T, k@pos^T: [8192,64] fp32 (BT = posB [K,D])
    dim3 gXP(K / 64, R / 64, 1);
    wmma_gemm<0><<<gXP, 128, 0, stream>>>(Qb, posB, xpq, R, K, D, 0, 0, 0);
    wmma_gemm<0><<<gXP, 128, 0, stream>>>(Kb, posB, xpk, R, K, D, 0, 0, 0);

    gaussian_q <<<blk(nXP), 256, 0, stream>>>(xpq, qn, p2, itv, amp, gqa, nXP);
    gaussian_kT<<<blk(nXP), 256, 0, stream>>>(xpk, kn, p2, itv, gkT, nXP, S);
    rowsum<<<blk(B * K), 256, 0, stream>>>(gkT, scol, B * K, S);

    // M[b] = gkT[b] @ V[b] : [64,768] = [64,2048] @ [2048,768], BT = VbT[b]
    dim3 gM(D / 64, K / 64, B);
    wmma_gemm<0><<<gM, 128, 0, stream>>>(gkT, VbT, Mf, K, D, S,
                                         (long long)K * S, (long long)D * S, (long long)K * D);
    transpose_f32_bf16<<<blk(B * K * D), 256, 0, stream>>>(Mf, MbT, B, K, D);

    // num[b] = gqa[b] @ M[b] : [2048,768] = [2048,64] @ [64,768], BT = MbT[b]
    dim3 gN(D / 64, S / 64, B);
    wmma_gemm<0><<<gN, 128, 0, stream>>>(gqa, MbT, Nmat, S, D, K,
                                         (long long)S * K, (long long)D * K, (long long)S * D);

    den_kernel <<<blk(R), 256, 0, stream>>>(gqa, scol, den, R, S);
    normalize_h<<<blk(nX), 256, 0, stream>>>(Nmat, den, hb, nX, D);

    // out = h @ Wo : [8192,768] fp32 -> d_out (BT = WoT)
    wmma_gemm<0><<<gProj, 128, 0, stream>>>(hb, woT, out, R, D, D, 0, 0, 0);
}
